// Gpt2MambaSelectiveTransformer_71829033059049
// MI455X (gfx1250) — matrix operand order repaired and checked by reading, not compiled, not run on hardware
//
#include <hip/hip_runtime.h>
#include <math.h>

// ---------------- model constants ----------------
#define L_LAYERS 12
#define HIDDEN   768
#define NUM_HEADS 12
#define HEAD_DIM  64
#define INTER    1536
#define STATE    16
#define DT_RANK  48
#define KCONV    4
#define VOCAB    50257
#define BATCH    2
#define SEQ      512
#define TOK      (BATCH*SEQ)          // 1024 rows
#define SSM_COLS (DT_RANK + 2*STATE)  // 80

#define NEG_INF (-__builtin_inff())

typedef _Float16 f16;
typedef __attribute__((ext_vector_type(4)))  _Float16 v4h;
typedef __attribute__((ext_vector_type(8)))  _Float16 v8h;
typedef __attribute__((ext_vector_type(16))) _Float16 v16h;
typedef __attribute__((ext_vector_type(8)))  float    v8f;

// ---------------- reductions (wave32) ----------------
__device__ inline float wave_sum(float v) {
#pragma unroll
  for (int o = 16; o > 0; o >>= 1) v += __shfl_down(v, o, 32);
  return v;
}
__device__ inline float wave_max(float v) {
#pragma unroll
  for (int o = 16; o > 0; o >>= 1) v = fmaxf(v, __shfl_down(v, o, 32));
  return v;
}
__device__ inline float block_sum(float v, float* red) {
  const int lane = threadIdx.x & 31, wid = threadIdx.x >> 5;
  v = wave_sum(v);
  if (lane == 0) red[wid] = v;
  __syncthreads();
  if (wid == 0) {
    float t = (lane < ((int)blockDim.x >> 5)) ? red[lane] : 0.f;
    t = wave_sum(t);
    if (lane == 0) red[0] = t;
  }
  __syncthreads();
  float r = red[0];
  __syncthreads();
  return r;
}
__device__ inline float block_max(float v, float* red) {
  const int lane = threadIdx.x & 31, wid = threadIdx.x >> 5;
  v = wave_max(v);
  if (lane == 0) red[wid] = v;
  __syncthreads();
  if (wid == 0) {
    float t = (lane < ((int)blockDim.x >> 5)) ? red[lane] : NEG_INF;
    t = wave_max(t);
    if (lane == 0) red[0] = t;
  }
  __syncthreads();
  float r = red[0];
  __syncthreads();
  return r;
}

// ---------------- generic WMMA GEMM ----------------
// C[z][M,N] = act(A[z][M,K] @ B + bias) (+ residual), f32 in/out, f16 WMMA compute.
// Block tile 64x128, 8 wave32 waves (4 M-stripes x 2 N-stripes), 4 accumulators/wave.
// 3-phase pipeline per K-step: issue next-tile global loads -> fragment ds_loads +
// 4x v_wmma on current LDS buffer (hides global latency) -> cvt+ds_store next tile.
#define BM  64
#define BN  128
#define BKT 32
#define LDT 40            // padded LDS row stride in halfs: 80B, 16B aligned, bank-spread
#define GTHREADS 256

constexpr int ACT_NONE = 0, ACT_GELU = 1, ACT_SOFTPLUS = 2;

template<int ACT, bool TRANSB>
__global__ __launch_bounds__(GTHREADS)
void k_gemm(const float* __restrict__ A, const float* __restrict__ W,
            const float* __restrict__ bias, const float* __restrict__ res,
            float* __restrict__ C, int M, int N, int K,
            int lda, int ldb, int ldc,
            long bsA, long bsB, long bsC)
{
  __shared__ f16 As[2][BM][LDT];   // [buf][m][k]  row-major: lane fragment = 2x b128
  __shared__ f16 Bt[2][BN][LDT];   // [buf][n][k]  transposed: lane fragment = 2x b128

  const int z = blockIdx.z;
  A += (size_t)z * bsA;
  W += (size_t)z * bsB;
  C += (size_t)z * bsC;

  const int tid  = threadIdx.x;
  const int lane = tid & 31;
  const int wid  = tid >> 5;
  const int wm   = wid >> 1;       // 0..3 : 16-row stripe
  const int wn   = wid & 1;        // 0..1 : 64-col stripe
  const int m0   = blockIdx.y * BM;
  const int n0   = blockIdx.x * BN;

  const bool fullM = (m0 + BM <= M);
  const bool fullN = (n0 + BN <= N);

  // ---- fast-path staging, split into load(regs) and store(LDS) phases ----
  auto loadA_fast = [&](int k0, float4* ra) {
#pragma unroll
    for (int it = 0; it < 2; ++it) {                 // 512 float4 chunks / 256 thr
      const int i = tid + it * GTHREADS;
      const int r = i >> 3, c4 = (i & 7) << 2;
      ra[it] = *reinterpret_cast<const float4*>(&A[(size_t)(m0 + r) * lda + k0 + c4]);
    }
  };
  auto storeA_fast = [&](const float4* ra, int buf) {
#pragma unroll
    for (int it = 0; it < 2; ++it) {
      const int i = tid + it * GTHREADS;
      const int r = i >> 3, c4 = (i & 7) << 2;
      v4h h = {(f16)ra[it].x, (f16)ra[it].y, (f16)ra[it].z, (f16)ra[it].w};
      *reinterpret_cast<v4h*>(&As[buf][r][c4]) = h;  // ds_store_b64
    }
  };
  auto loadB_fast = [&](int k0, float4* rb) {
#pragma unroll
    for (int it = 0; it < 4; ++it) {
      const int i = tid + it * GTHREADS;
      if (TRANSB) {                                  // W is [N][K]: contiguous in k
        const int n = i >> 3, k4 = (i & 7) << 2;
        rb[it] = *reinterpret_cast<const float4*>(&W[(size_t)(n0 + n) * ldb + k0 + k4]);
      } else {                                       // W is [K][N]: coalesced in n
        // scalar-quad loads: no 16B-alignment requirement (ldb may be odd, e.g. 50257)
        const int r = i >> 5, c4 = (i & 31) << 2;
        const float* src = &W[(size_t)(k0 + r) * ldb + n0 + c4];
        rb[it].x = src[0]; rb[it].y = src[1]; rb[it].z = src[2]; rb[it].w = src[3];
      }
    }
  };
  auto storeB_fast = [&](const float4* rb, int buf) {
#pragma unroll
    for (int it = 0; it < 4; ++it) {
      const int i = tid + it * GTHREADS;
      if (TRANSB) {
        const int n = i >> 3, k4 = (i & 7) << 2;
        v4h h = {(f16)rb[it].x, (f16)rb[it].y, (f16)rb[it].z, (f16)rb[it].w};
        *reinterpret_cast<v4h*>(&Bt[buf][n][k4]) = h;
      } else {
        const int r = i >> 5, c4 = (i & 31) << 2;
        Bt[buf][c4 + 0][r] = (f16)rb[it].x;
        Bt[buf][c4 + 1][r] = (f16)rb[it].y;
        Bt[buf][c4 + 2][r] = (f16)rb[it].z;
        Bt[buf][c4 + 3][r] = (f16)rb[it].w;
      }
    }
  };

  // ---- branchless edge staging (rare: K=48 pad, N tails) ----
  auto stageA_edge = [&](int k0, int buf) {
    for (int i = tid; i < BM * BKT; i += GTHREADS) {
      const int r = i >> 5, c = i & 31;
      const int gm = m0 + r, gk = k0 + c;
      const bool ok = (gm < M) && (gk < K);
      const float v = A[ok ? ((size_t)gm * lda + gk) : 0];
      As[buf][r][c] = (f16)(ok ? v : 0.f);
    }
  };
  auto stageB_edge = [&](int k0, int buf) {
    for (int i = tid; i < BKT * BN; i += GTHREADS) {
      const int r = i >> 7, c = i & 127;             // r = k, c = n
      const int gk = k0 + r, gn = n0 + c;
      const bool ok = (gk < K) && (gn < N);
      const float v = TRANSB ? W[ok ? ((size_t)gn * ldb + gk) : 0]
                             : W[ok ? ((size_t)gk * ldb + gn) : 0];
      Bt[buf][c][r] = (f16)(ok ? v : 0.f);
    }
  };

  v8f acc[4] = {};

  // ---- prologue: stage tile 0 ----
  {
    const bool fK0 = (BKT <= K);
    if (fullM && fK0) { float4 ra[2]; loadA_fast(0, ra); storeA_fast(ra, 0); }
    else              stageA_edge(0, 0);
    if (fullN && fK0) { float4 rb[4]; loadB_fast(0, rb); storeB_fast(rb, 0); }
    else              stageB_edge(0, 0);
  }
  __syncthreads();

  int cur = 0;
  for (int k0 = 0; k0 < K; k0 += BKT) {
    const int  k0n    = k0 + BKT;
    const bool have   = (k0n < K);
    const bool fKn    = (k0n + BKT <= K);
    const bool fastAn = fullM && fKn;
    const bool fastBn = fullN && fKn;

    // phase 1: issue next tile's global loads into registers
    float4 ra[2], rb[4];
    if (have && fastAn) loadA_fast(k0n, ra);
    if (have && fastBn) loadB_fast(k0n, rb);

    // phase 2: fragments (all ds_loads first) + 4 back-to-back WMMAs
    const int ar = wm * 16 + (lane & 15);
    const int ak = (lane >> 4) << 3;                 // +8 halfs for lanes 16..31
    v8h alo = *reinterpret_cast<const v8h*>(&As[cur][ar][ak]);
    v8h ahi = *reinterpret_cast<const v8h*>(&As[cur][ar][ak + 16]);
    v16h afrag = __builtin_shufflevector(alo, ahi,
        0,1,2,3,4,5,6,7,8,9,10,11,12,13,14,15);

    const int bn = wn * 64 + (lane & 15);
    const int bk = (lane >> 4) << 3;                 // +8 halfs for lanes 16..31
    v16h bfrag[4];
#pragma unroll
    for (int tn = 0; tn < 4; ++tn) {
      v8h blo = *reinterpret_cast<const v8h*>(&Bt[cur][bn + tn * 16][bk]);
      v8h bhi = *reinterpret_cast<const v8h*>(&Bt[cur][bn + tn * 16][bk + 16]);
      bfrag[tn] = __builtin_shufflevector(blo, bhi,
          0,1,2,3,4,5,6,7,8,9,10,11,12,13,14,15);
    }
#pragma unroll
    for (int tn = 0; tn < 4; ++tn)
      acc[tn] = __builtin_amdgcn_wmma_f32_16x16x32_f16(false, afrag, false, bfrag[tn],
                                                       (short)0, acc[tn], false, false);

    // phase 3: convert + store next tile into the other buffer
    if (have) {
      if (fastAn) storeA_fast(ra, cur ^ 1); else stageA_edge(k0n, cur ^ 1);
      if (fastBn) storeB_fast(rb, cur ^ 1); else stageB_edge(k0n, cur ^ 1);
    }
    __syncthreads();   // next tile staged & current tile consumed
    cur ^= 1;
  }

  // epilogue: C/D layout -> row = e + 8*(lane/16), col = lane%16
#pragma unroll
  for (int tn = 0; tn < 4; ++tn) {
    const int col = n0 + wn * 64 + tn * 16 + (lane & 15);
#pragma unroll
    for (int e = 0; e < 8; ++e) {
      const int row = m0 + wm * 16 + e + ((lane >> 4) << 3);
      if (row < M && col < N) {
        float val = acc[tn][e];
        if (bias) val += bias[col];
        if (ACT == ACT_GELU)          val = 0.5f * val * (1.f + erff(val * 0.70710678118f));
        else if (ACT == ACT_SOFTPLUS) val = (val > 20.f) ? val : log1pf(expf(val));
        if (res) val += res[(size_t)row * ldc + col];
        C[(size_t)row * ldc + col] = val;
      }
    }
  }
}

static void gemm(hipStream_t st, const float* A, const float* B, const float* bias,
                 const float* res, float* C, int M, int N, int K,
                 int lda, int ldb, int ldc, int act, bool transB,
                 int batch = 1, long bsA = 0, long bsB = 0, long bsC = 0)
{
  dim3 grid((N + BN - 1) / BN, (M + BM - 1) / BM, batch);
  dim3 block(GTHREADS);
  if (transB)
    k_gemm<ACT_NONE, true ><<<grid, block, 0, st>>>(A, B, bias, res, C, M, N, K, lda, ldb, ldc, bsA, bsB, bsC);
  else if (act == ACT_GELU)
    k_gemm<ACT_GELU, false><<<grid, block, 0, st>>>(A, B, bias, res, C, M, N, K, lda, ldb, ldc, bsA, bsB, bsC);
  else if (act == ACT_SOFTPLUS)
    k_gemm<ACT_SOFTPLUS, false><<<grid, block, 0, st>>>(A, B, bias, res, C, M, N, K, lda, ldb, ldc, bsA, bsB, bsC);
  else
    k_gemm<ACT_NONE, false><<<grid, block, 0, st>>>(A, B, bias, res, C, M, N, K, lda, ldb, ldc, bsA, bsB, bsC);
}

// ---------------- elementwise / reduction kernels ----------------
__global__ void k_embed(const int* __restrict__ ids, const float* __restrict__ emb,
                        const float* __restrict__ pos, float* __restrict__ x)
{
  const int row = blockIdx.x;          // 0..TOK-1
  const int t   = row % SEQ;
  const int id  = ids[row];
  for (int j = threadIdx.x; j < HIDDEN; j += blockDim.x)
    x[(size_t)row * HIDDEN + j] = emb[(size_t)id * HIDDEN + j] + pos[(size_t)t * HIDDEN + j];
}

__global__ void k_layernorm(const float* __restrict__ x, const float* __restrict__ w,
                            const float* __restrict__ b, float* __restrict__ y)
{
  __shared__ float red[8];
  const int row = blockIdx.x;
  const float* xr = x + (size_t)row * HIDDEN;
  float s = 0.f;
  for (int j = threadIdx.x; j < HIDDEN; j += blockDim.x) s += xr[j];
  const float mu = block_sum(s, red) * (1.f / HIDDEN);
  float v = 0.f;
  for (int j = threadIdx.x; j < HIDDEN; j += blockDim.x) { float d = xr[j] - mu; v += d * d; }
  const float var = block_sum(v, red) * (1.f / HIDDEN);
  const float inv = rsqrtf(var + 1e-5f);
  for (int j = threadIdx.x; j < HIDDEN; j += blockDim.x)
    y[(size_t)row * HIDDEN + j] = (xr[j] - mu) * inv * w[j] + b[j];
}

__global__ void k_rmsnorm(const float* __restrict__ x, const float* __restrict__ w,
                          float* __restrict__ y)
{
  __shared__ float red[8];
  const int row = blockIdx.x;
  const float* xr = x + (size_t)row * HIDDEN;
  float v = 0.f;
  for (int j = threadIdx.x; j < HIDDEN; j += blockDim.x) v += xr[j] * xr[j];
  const float var = block_sum(v, red) * (1.f / HIDDEN);
  const float inv = rsqrtf(var + 1e-5f);
  for (int j = threadIdx.x; j < HIDDEN; j += blockDim.x)
    y[(size_t)row * HIDDEN + j] = xr[j] * inv * w[j];
}

// u = silu(conv_b + causal depthwise conv over hs = proj[..., :INTER])
__global__ void k_conv_silu(const float* __restrict__ proj, const float* __restrict__ cw,
                            const float* __restrict__ cb, float* __restrict__ u, int total)
{
  const int idx = blockIdx.x * blockDim.x + threadIdx.x;
  if (idx >= total) return;
  const int i  = idx % INTER;
  const int bt = idx / INTER;
  const int t  = bt % SEQ;
  float acc = cb[i];
#pragma unroll
  for (int k = 0; k < KCONV; ++k) {
    const int ts = t + k - (KCONV - 1);
    if (ts >= 0)
      acc += proj[(size_t)(bt - t + ts) * (2 * INTER) + i] * cw[i * KCONV + k];
  }
  u[idx] = acc / (1.f + expf(-acc));
}

// Sequential selective-scan over T; one thread per (b, i). Gate SiLU fused.
__global__ void k_scan(const float* __restrict__ dt, const float* __restrict__ u,
                       const float* __restrict__ ssm, const float* __restrict__ proj,
                       const float* __restrict__ Alog, const float* __restrict__ D,
                       float* __restrict__ y)
{
  const int idx = blockIdx.x * blockDim.x + threadIdx.x;
  if (idx >= BATCH * INTER) return;
  const int b = idx / INTER, i = idx % INTER;

  float Arow[STATE];
#pragma unroll
  for (int s = 0; s < STATE; ++s) Arow[s] = -expf(Alog[i * STATE + s]);
  float st[STATE];
#pragma unroll
  for (int s = 0; s < STATE; ++s) st[s] = 0.f;
  const float Dv = D[i];

  for (int t = 0; t < SEQ; ++t) {
    const size_t base = (size_t)(b * SEQ + t);
    const float dtv = dt[base * INTER + i];
    const float uv  = u [base * INTER + i];
    const float* bc = ssm + base * SSM_COLS;
    float yv = 0.f;
#pragma unroll
    for (int s = 0; s < STATE; ++s) {
      st[s] = expf(dtv * Arow[s]) * st[s] + dtv * bc[DT_RANK + s] * uv;
      yv += st[s] * bc[DT_RANK + STATE + s];
    }
    yv += uv * Dv;
    const float g = proj[base * (2 * INTER) + INTER + i];
    yv *= g / (1.f + expf(-g));
    y[base * INTER + i] = yv;
  }
}

// sel = sigmoid(clip(m,-20,20) . sel_w + sel_b), one block per token
__global__ void k_sel(const float* __restrict__ m, const float* __restrict__ sw,
                      const float* __restrict__ sb, float* __restrict__ sel)
{
  __shared__ float red[8];
  const int row = blockIdx.x;
  float s = 0.f;
  for (int j = threadIdx.x; j < HIDDEN; j += blockDim.x) {
    float mv = m[(size_t)row * HIDDEN + j];
    mv = fminf(20.f, fmaxf(-20.f, mv));
    s += mv * sw[j];
  }
  const float d = block_sum(s, red) + sb[0];
  if (threadIdx.x == 0) sel[row] = 1.f / (1.f + expf(-d));
}

// split qkv [B,T,3H] -> q/k/v in [B,H,T,hd]; V pre-scaled by (0.3 + 0.7*sel[b,t])
__global__ void k_split_qkv(const float* __restrict__ qkv, const float* __restrict__ sel,
                            float* __restrict__ q, float* __restrict__ k,
                            float* __restrict__ v, int total)
{
  const int idx = blockIdx.x * blockDim.x + threadIdx.x;
  if (idx >= total) return;
  const int c  = idx % HIDDEN;
  const int bt = idx / HIDDEN;
  const int b  = bt / SEQ, t = bt % SEQ;
  const int h  = c / HEAD_DIM, d = c % HEAD_DIM;
  const size_t src = (size_t)bt * (3 * HIDDEN);
  const size_t dst = (((size_t)(b * NUM_HEADS + h) * SEQ) + t) * HEAD_DIM + d;
  q[dst] = qkv[src + c];
  k[dst] = qkv[src + HIDDEN + c];
  v[dst] = qkv[src + 2 * HIDDEN + c] * (0.3f + 0.7f * sel[bt]);
}

// per-row softmax over keys with scale + causal + pad mask; in place
__global__ void k_softmax_mask(float* __restrict__ scores, const int* __restrict__ amask,
                               float scale)
{
  __shared__ float red[8];
  const int row = blockIdx.x;             // (b*H + h)*SEQ + t
  const int t   = row % SEQ;
  const int bh  = row / SEQ;
  const int b   = bh / NUM_HEADS;
  float* sr = scores + (size_t)row * SEQ;

  float mx = NEG_INF;
  for (int j = threadIdx.x; j < SEQ; j += blockDim.x) {
    float v = sr[j] * scale;
    if (j > t || amask[b * SEQ + j] == 0) v = NEG_INF;
    sr[j] = v;
    mx = fmaxf(mx, v);
  }
  mx = block_max(mx, red);
  if (!(mx > NEG_INF)) {                  // fully-masked row -> nan_to_num -> zeros
    for (int j = threadIdx.x; j < SEQ; j += blockDim.x) sr[j] = 0.f;
    return;
  }
  float s = 0.f;
  for (int j = threadIdx.x; j < SEQ; j += blockDim.x) {
    const float e = expf(sr[j] - mx);
    sr[j] = e;
    s += e;
  }
  s = block_sum(s, red);
  const float inv = 1.f / s;
  for (int j = threadIdx.x; j < SEQ; j += blockDim.x) sr[j] *= inv;
}

// [B,H,T,hd] -> [B,T,H*hd]
__global__ void k_merge(const float* __restrict__ at, float* __restrict__ merged, int total)
{
  const int idx = blockIdx.x * blockDim.x + threadIdx.x;
  if (idx >= total) return;
  const int c  = idx % HIDDEN;
  const int bt = idx / HIDDEN;
  const int b  = bt / SEQ, t = bt % SEQ;
  const int h  = c / HEAD_DIM, d = c % HEAD_DIM;
  merged[idx] = at[(((size_t)(b * NUM_HEADS + h) * SEQ) + t) * HEAD_DIM + d];
}

// ---------------- host orchestration ----------------
extern "C" void kernel_launch(void* const* d_in, const int* in_sizes, int n_in,
                              void* d_out, int out_size, void* d_ws, size_t ws_size,
                              hipStream_t stream)
{
  (void)in_sizes; (void)n_in; (void)out_size; (void)ws_size;

  // setup_inputs() dict-insertion order
  const int*   ids   = (const int*)  d_in[0];
  const int*   amask = (const int*)  d_in[1];
  const float* embed = (const float*)d_in[2];
  const float* pos   = (const float*)d_in[3];
  const float* lnf_w = (const float*)d_in[4];
  const float* lnf_b = (const float*)d_in[5];
  const float* lmw   = (const float*)d_in[6];
  // layers dict order (each leading dim L)
  const float* P_ln1_w    = (const float*)d_in[7];
  const float* P_ln1_b    = (const float*)d_in[8];
  const float* P_ln2_w    = (const float*)d_in[9];
  const float* P_ln2_b    = (const float*)d_in[10];
  const float* P_mln_w    = (const float*)d_in[11];
  const float* P_mln_b    = (const float*)d_in[12];
  const float* P_mnorm_w  = (const float*)d_in[13];
  const float* P_cattn_w  = (const float*)d_in[14];
  const float* P_cattn_b  = (const float*)d_in[15];
  const float* P_cproj_w  = (const float*)d_in[16];
  const float* P_cproj_b  = (const float*)d_in[17];
  const float* P_inproj_w = (const float*)d_in[18];
  const float* P_conv_w   = (const float*)d_in[19];
  const float* P_conv_b   = (const float*)d_in[20];
  const float* P_xproj_w  = (const float*)d_in[21];
  const float* P_dtproj_w = (const float*)d_in[22];
  const float* P_dtproj_b = (const float*)d_in[23];
  const float* P_Alog     = (const float*)d_in[24];
  const float* P_D        = (const float*)d_in[25];
  const float* P_outproj_w= (const float*)d_in[26];
  const float* P_sel_w    = (const float*)d_in[27];
  const float* P_sel_b    = (const float*)d_in[28];
  const float* P_mlp_w1   = (const float*)d_in[29];
  const float* P_mlp_b1   = (const float*)d_in[30];
  const float* P_mlp_w2   = (const float*)d_in[31];
  const float* P_mlp_b2   = (const float*)d_in[32];

  // workspace carve-out (f32), with deliberate reuse
  float* ws = (float*)d_ws;
  size_t off = 0;
  auto alloc = [&](size_t n) { float* p = ws + off; off += (n + 255) & ~(size_t)255; return p; };
  float* x    = alloc((size_t)TOK * HIDDEN);
  float* lnm  = alloc((size_t)TOK * HIDDEN);          // mamba-LN(x); residual for mixer
  float* bufa = alloc((size_t)TOK * HIDDEN);          // rms -> m -> merged
  float* proj = alloc((size_t)TOK * 2 * INTER);       // mamba proj; later MLP hidden
  float* u    = alloc((size_t)TOK * INTER);
  float* xn   = alloc((size_t)TOK * HIDDEN);          // LN outputs (ln1/ln2/final)
  float* ssm  = alloc((size_t)TOK * SSM_COLS);
  float* dty  = alloc((size_t)TOK * INTER);           // dt, then y (in-place in scan)
  float* sel  = alloc((size_t)TOK);
  float* qkv  = alloc((size_t)TOK * 3 * HIDDEN);
  float* qT   = alloc((size_t)TOK * HIDDEN);
  float* kT   = alloc((size_t)TOK * HIDDEN);
  float* vT   = alloc((size_t)TOK * HIDDEN);
  float* sc   = alloc((size_t)BATCH * NUM_HEADS * SEQ * SEQ);
  float* aT   = alloc((size_t)TOK * HIDDEN);

  const int EW = 256;
  k_embed<<<TOK, EW, 0, stream>>>(ids, embed, pos, x);

  for (int l = 0; l < L_LAYERS; ++l) {
    const float* ln1w   = P_ln1_w    + (size_t)l * HIDDEN;
    const float* ln1b   = P_ln1_b    + (size_t)l * HIDDEN;
    const float* ln2w   = P_ln2_w    + (size_t)l * HIDDEN;
    const float* ln2b   = P_ln2_b    + (size_t)l * HIDDEN;
    const float* mlnw   = P_mln_w    + (size_t)l * HIDDEN;
    const float* mlnb   = P_mln_b    + (size_t)l * HIDDEN;
    const float* mnormw = P_mnorm_w  + (size_t)l * HIDDEN;
    const float* cattnw = P_cattn_w  + (size_t)l * HIDDEN * 3 * HIDDEN;
    const float* cattnb = P_cattn_b  + (size_t)l * 3 * HIDDEN;
    const float* cprojw = P_cproj_w  + (size_t)l * HIDDEN * HIDDEN;
    const float* cprojb = P_cproj_b  + (size_t)l * HIDDEN;
    const float* inprojw= P_inproj_w + (size_t)l * HIDDEN * 2 * INTER;
    const float* convw  = P_conv_w   + (size_t)l * INTER * KCONV;
    const float* convb  = P_conv_b   + (size_t)l * INTER;
    const float* xprojw = P_xproj_w  + (size_t)l * INTER * SSM_COLS;
    const float* dtprojw= P_dtproj_w + (size_t)l * DT_RANK * INTER;
    const float* dtprojb= P_dtproj_b + (size_t)l * INTER;
    const float* Alog   = P_Alog     + (size_t)l * INTER * STATE;
    const float* Dssm   = P_D        + (size_t)l * INTER;
    const float* outpw  = P_outproj_w+ (size_t)l * INTER * HIDDEN;
    const float* selw   = P_sel_w    + (size_t)l * HIDDEN;
    const float* selb   = P_sel_b    + (size_t)l;
    const float* w1     = P_mlp_w1   + (size_t)l * HIDDEN * 4 * HIDDEN;
    const float* b1     = P_mlp_b1   + (size_t)l * 4 * HIDDEN;
    const float* w2     = P_mlp_w2   + (size_t)l * 4 * HIDDEN * HIDDEN;
    const float* b2     = P_mlp_b2   + (size_t)l * HIDDEN;

    // ---- mamba path (produces sel gate for V) ----
    k_layernorm<<<TOK, EW, 0, stream>>>(x, mlnw, mlnb, lnm);
    k_rmsnorm  <<<TOK, EW, 0, stream>>>(lnm, mnormw, bufa);
    gemm(stream, bufa, inprojw, nullptr, nullptr, proj,
         TOK, 2 * INTER, HIDDEN, HIDDEN, 2 * INTER, 2 * INTER, ACT_NONE, false);
    k_conv_silu<<<(TOK * INTER + EW - 1) / EW, EW, 0, stream>>>(proj, convw, convb, u, TOK * INTER);
    gemm(stream, u, xprojw, nullptr, nullptr, ssm,
         TOK, SSM_COLS, INTER, INTER, SSM_COLS, SSM_COLS, ACT_NONE, false);
    gemm(stream, ssm, dtprojw, dtprojb, nullptr, dty,            // A = ssm[:, :48], lda=80
         TOK, INTER, DT_RANK, SSM_COLS, INTER, INTER, ACT_SOFTPLUS, false);
    k_scan<<<(BATCH * INTER + EW - 1) / EW, EW, 0, stream>>>(dty, u, ssm, proj, Alog, Dssm, dty);
    gemm(stream, dty, outpw, nullptr, lnm, bufa,                 // m = ln_m + y @ out_proj
         TOK, HIDDEN, INTER, INTER, HIDDEN, HIDDEN, ACT_NONE, false);
    k_sel<<<TOK, EW, 0, stream>>>(bufa, selw, selb, sel);

    // ---- attention ----
    k_layernorm<<<TOK, EW, 0, stream>>>(x, ln1w, ln1b, xn);
    gemm(stream, xn, cattnw, cattnb, nullptr, qkv,
         TOK, 3 * HIDDEN, HIDDEN, HIDDEN, 3 * HIDDEN, 3 * HIDDEN, ACT_NONE, false);
    k_split_qkv<<<(TOK * HIDDEN + EW - 1) / EW, EW, 0, stream>>>(qkv, sel, qT, kT, vT, TOK * HIDDEN);
    // scores[bh] = Q[bh] @ K[bh]^T   (transB), batched over B*H
    gemm(stream, qT, kT, nullptr, nullptr, sc,
         SEQ, SEQ, HEAD_DIM, HEAD_DIM, HEAD_DIM, SEQ, ACT_NONE, true,
         BATCH * NUM_HEADS, (long)SEQ * HEAD_DIM, (long)SEQ * HEAD_DIM, (long)SEQ * SEQ);
    k_softmax_mask<<<BATCH * NUM_HEADS * SEQ, EW, 0, stream>>>(sc, amask, 0.125f);
    // out[bh] = P[bh] @ V[bh]
    gemm(stream, sc, vT, nullptr, nullptr, aT,
         SEQ, HEAD_DIM, SEQ, SEQ, HEAD_DIM, HEAD_DIM, ACT_NONE, false,
         BATCH * NUM_HEADS, (long)SEQ * SEQ, (long)SEQ * HEAD_DIM, (long)SEQ * HEAD_DIM);
    k_merge<<<(TOK * HIDDEN + EW - 1) / EW, EW, 0, stream>>>(aT, bufa, TOK * HIDDEN);
    gemm(stream, bufa, cprojw, cprojb, x, x,                     // x += out @ c_proj + b
         TOK, HIDDEN, HIDDEN, HIDDEN, HIDDEN, HIDDEN, ACT_NONE, false);

    // ---- MLP ----
    k_layernorm<<<TOK, EW, 0, stream>>>(x, ln2w, ln2b, xn);
    gemm(stream, xn, w1, b1, nullptr, proj,
         TOK, 4 * HIDDEN, HIDDEN, HIDDEN, 4 * HIDDEN, 4 * HIDDEN, ACT_GELU, false);
    gemm(stream, proj, w2, b2, x, x,
         TOK, HIDDEN, 4 * HIDDEN, 4 * HIDDEN, HIDDEN, HIDDEN, ACT_NONE, false);
  }

  // final LN + lm_head (154MB weight is L2-resident on the 192MB MI455X L2)
  k_layernorm<<<TOK, EW, 0, stream>>>(x, lnf_w, lnf_b, xn);
  gemm(stream, xn, lmw, nullptr, nullptr, (float*)d_out,
       TOK, VOCAB, HIDDEN, HIDDEN, VOCAB, VOCAB, ACT_NONE, false);
}
